// AttentionOperator_70394513981644
// MI455X (gfx1250) — compile-verified, matrix operand order checked by reading, not run on hardware
//
#include <hip/hip_runtime.h>
#include <hip/hip_bf16.h>

typedef __attribute__((ext_vector_type(16))) _Float16 v16h;
typedef __attribute__((ext_vector_type(8)))  float    v8f;

#define PI2 6.28318530717958647692f

// ---------------- WMMA tile helpers (gfx1250 wave32, 16x16x32 f16) ----------
// A (16x32 f16): lane L<16 -> row L, K {0..7,16..23}; lane>=16 -> K {8..15,24..31}
__device__ __forceinline__ v16h ld_tile_h16(const _Float16* base, int ld) {
  int lane = threadIdx.x & 31;
  int r  = lane & 15;
  int kb = (lane & 16) ? 8 : 0;
  const _Float16* p = base + (size_t)r * ld + kb;
  v16h a;
#pragma unroll
  for (int i = 0; i < 8; ++i) { a[i] = p[i]; a[8 + i] = p[16 + i]; }
  return a;
}

__device__ __forceinline__ v16h ld_tile_f32cvt(const float* base, int ld) {
  int lane = threadIdx.x & 31;
  int r  = lane & 15;
  int kb = (lane & 16) ? 8 : 0;
  const float* p = base + (size_t)r * ld + kb;
  v16h a;
#pragma unroll
  for (int i = 0; i < 8; ++i) { a[i] = (_Float16)p[i]; a[8 + i] = (_Float16)p[16 + i]; }
  return a;
}

__device__ __forceinline__ v8f wmma_f16(v16h a, v16h b, v8f c) {
  return __builtin_amdgcn_wmma_f32_16x16x32_f16(false, a, false, b, (short)0, c,
                                                false, false);
}

// D (16x16 f32): lane = N + 16*(M>=8); VGPR v holds M = v + 8*(lane>=16)
__device__ __forceinline__ void st_tile_f32(float* base, int ld, v8f d) {
  int lane = threadIdx.x & 31;
  int n  = lane & 15;
  int mb = (lane & 16) ? 8 : 0;
#pragma unroll
  for (int v = 0; v < 8; ++v) base[(size_t)(mb + v) * ld + n] = d[v];
}

__device__ __forceinline__ void st_tile_f16(_Float16* base, int ld, v8f d) {
  int lane = threadIdx.x & 31;
  int n  = lane & 15;
  int mb = (lane & 16) ? 8 : 0;
#pragma unroll
  for (int v = 0; v < 8; ++v) base[(size_t)(mb + v) * ld + n] = (_Float16)d[v];
}

// ---------------- Kernel 0: DFT basis tables --------------------------------
// BwT[c][w]   (32x128 f16): c<16 -> cos(2pi w c/128), c>=16 -> -sin(2pi w (c-16)/128)
// Ec/Es[k][h] (32x128 f32): twiddles for restricted h modes (0..15, 112..127)
// BinvT[n][c] (128x32 f16): irfft-w basis with c_k doubling and 1/128
__global__ void fno_basis(_Float16* BwT, float* Ec, float* Es, _Float16* BinvT) {
  int t = blockIdx.x * 128 + threadIdx.x;  // 4096 threads
  {
    int c = t >> 7, w = t & 127;
    float v = (c < 16) ? cosf(PI2 * (float)(w * c) / 128.f)
                       : -sinf(PI2 * (float)(w * (c - 16)) / 128.f);
    BwT[t] = (_Float16)v;
  }
  {
    int k = t >> 7, h = t & 127;
    int mode = (k < 16) ? k : (96 + k);     // 112..127 for k=16..31
    float ang = PI2 * (float)(h * mode) / 128.f;
    Ec[t] = cosf(ang);
    Es[t] = sinf(ang);
  }
  {
    int n = t >> 5, c = t & 31;
    float v;
    if (c < 16) {
      float sc = (c == 0 ? 1.f : 2.f) / 128.f;
      v = sc * cosf(PI2 * (float)(n * c) / 128.f);
    } else {
      int ky = c - 16;
      float sc = (ky == 0 ? 1.f : 2.f) / 128.f;
      v = -sc * sinf(PI2 * (float)(n * ky) / 128.f);
    }
    BinvT[t] = (_Float16)v;
  }
}

// ---------------- Kernel Z: zero the x-half of output -----------------------
__global__ void fno_zero(float* out) {
  int tid = blockIdx.x * 256 + threadIdx.x;       // 4*32*16384 threads
  int bt = tid >> 14, i = tid & 16383;
  out[(size_t)bt * 32768 + i] = 0.f;
}

// ---------------- Kernel 1: forward w-DFT (WMMA) ----------------------------
// F1[frame][h][c] = X(128x128) * BwT^T ; frames 0..127 = x half, 128..255 = y half
__global__ void fno_fwdw(const float* __restrict__ z, const _Float16* __restrict__ BwT,
                         float* __restrict__ F1x, float* __restrict__ F1y) {
  int fid = blockIdx.x;             // 0..255
  int wv  = threadIdx.x >> 5;       // 4 waves
  const float* src; float* dst;
  if (fid < 128) { src = z + (size_t)fid * 32768;          dst = F1x + (size_t)fid * 4096; }
  else           { src = z + (size_t)(fid - 128) * 32768 + 16384;
                   dst = F1y + (size_t)(fid - 128) * 4096; }
#pragma unroll
  for (int rep = 0; rep < 2; ++rep) {
    int hb = wv + rep * 4;          // 8 h-blocks
#pragma unroll
    for (int cb = 0; cb < 2; ++cb) {
      v8f acc = {};
#pragma unroll
      for (int kc = 0; kc < 4; ++kc) {
        v16h a = ld_tile_f32cvt(src + hb * 16 * 128 + kc * 32, 128);
        v16h b = ld_tile_h16(BwT + cb * 16 * 128 + kc * 32, 128);
        acc = wmma_f16(a, b, acc);
      }
      st_tile_f32(dst + hb * 16 * 32 + cb * 16, 32, acc);
    }
  }
}

// ---------------- Kernel 2: restricted h-DFT + weights + inverse h (VALU) ---
__global__ void fno_modes(const float* __restrict__ F1x, const float* __restrict__ F1y,
                          const float* qw1r, const float* qw1i,
                          const float* qw2r, const float* qw2i,
                          const float* kw1r, const float* kw1i,
                          const float* kw2r, const float* kw2i,
                          const float* icl,
                          const float* __restrict__ Ec, const float* __restrict__ Es,
                          _Float16* Gq, _Float16* Gk, _Float16* Gv) {
  __shared__ float s_f1x[128 * 32];
  __shared__ float s_f1y[128 * 32];
  __shared__ float s_f2r[3 * 512];
  __shared__ float s_f2i[3 * 512];
  int f = blockIdx.x, tid = threadIdx.x;   // 256 threads
  const float* fx = F1x + (size_t)f * 4096;
  const float* fy = F1y + (size_t)f * 4096;
  for (int i = tid; i < 4096; i += 256) { s_f1x[i] = fx[i]; s_f1y[i] = fy[i]; }
  __syncthreads();
  float sc = icl[0];
  for (int e = tid; e < 1536; e += 256) {
    int set = e >> 9;             // 0=q, 1=k, 2=v
    int rem = e & 511;
    int kxi = rem >> 4, ky = rem & 15;
    const float* F1 = (set == 2) ? s_f1y : s_f1x;
    const float* ec = Ec + kxi * 128;
    const float* es = Es + kxi * 128;
    float fr = 0.f, fi = 0.f;
    for (int h = 0; h < 128; ++h) {
      float r = F1[h * 32 + ky], im = F1[h * 32 + 16 + ky];
      float c = ec[h], s = es[h];
      fr += r * c + im * s;       // (r+i*im)*(c - i*s)
      fi += im * c - r * s;
    }
    float wr, wi;
    if (set == 0) {
      wr = (kxi < 16) ? qw1r[kxi * 16 + ky] : qw2r[(kxi - 16) * 16 + ky];
      wi = (kxi < 16) ? qw1i[kxi * 16 + ky] : qw2i[(kxi - 16) * 16 + ky];
    } else if (set == 1) {
      wr = (kxi < 16) ? kw1r[kxi * 16 + ky] : kw2r[(kxi - 16) * 16 + ky];
      wi = (kxi < 16) ? kw1i[kxi * 16 + ky] : kw2i[(kxi - 16) * 16 + ky];
    } else { wr = sc; wi = 0.f; }
    s_f2r[e] = wr * fr - wi * fi;
    s_f2i[e] = wr * fi + wi * fr;
  }
  __syncthreads();
  for (int e = tid; e < 3 * 2048; e += 256) {
    int set = e >> 11;
    int rem = e & 2047;
    int m = rem >> 4, ky = rem & 15;
    const float* f2r = s_f2r + set * 512;
    const float* f2i = s_f2i + set * 512;
    float gr = 0.f, gi = 0.f;
#pragma unroll
    for (int k = 0; k < 32; ++k) {
      float c = Ec[k * 128 + m], s = Es[k * 128 + m];
      float r = f2r[k * 16 + ky], im = f2i[k * 16 + ky];
      gr += c * r - s * im;       // e^{+i ang} * F2w
      gi += c * im + s * r;
    }
    gr *= (1.f / 128.f); gi *= (1.f / 128.f);
    _Float16* G = (set == 0) ? Gq : ((set == 1) ? Gk : Gv);
    G[(size_t)f * 4096 + m * 32 + ky]      = (_Float16)gr;
    G[(size_t)f * 4096 + m * 32 + 16 + ky] = (_Float16)gi;
  }
}

// ---------------- Kernel 3: inverse w-DFT (WMMA) ----------------------------
// set 0/1 -> f16 rows of M = [Q;K] per batch; set 2 -> f32 value frames
__global__ void fno_invw(const _Float16* __restrict__ Gq, const _Float16* __restrict__ Gk,
                         const _Float16* __restrict__ Gv, const _Float16* __restrict__ BinvT,
                         _Float16* __restrict__ M, float* __restrict__ value) {
  int f = blockIdx.x;           // 0..127
  int set = blockIdx.y;         // 0..2
  int wv = threadIdx.x >> 5;
  const _Float16* G = ((set == 0) ? Gq : ((set == 1) ? Gk : Gv)) + (size_t)f * 4096;
  int b = f >> 5, t = f & 31;
  for (int tile = wv; tile < 64; tile += 4) {
    int hb = tile >> 3, nb = tile & 7;
    v16h a  = ld_tile_h16(G + hb * 16 * 32, 32);
    v16h bb = ld_tile_h16(BinvT + nb * 16 * 32, 32);
    v8f acc = {};
    acc = wmma_f16(a, bb, acc);
    if (set < 2) {
      _Float16* dstp = M + ((size_t)(b * 64 + set * 32 + t)) * 16384 + hb * 16 * 128 + nb * 16;
      st_tile_f16(dstp, 128, acc);
    } else {
      float* dstp = value + (size_t)f * 16384 + hb * 16 * 128 + nb * 16;
      st_tile_f32(dstp, 128, acc);
    }
  }
}

// ---------------- Kernel 4: 5x5 circular conv, write transposed f16 ---------
__global__ void fno_conv(const float* __restrict__ value, const float* __restrict__ kk,
                         _Float16* __restrict__ vcT) {
  int tid = blockIdx.x * 256 + threadIdx.x;   // 128*16384 threads
  int f = tid >> 14, p = tid & 16383;
  int h = p >> 7, w = p & 127;
  const float* vf = value + (size_t)f * 16384;
  float acc = 0.f;
#pragma unroll
  for (int dy = 0; dy < 5; ++dy) {
    int hh = (h + dy + 126) & 127;
#pragma unroll
    for (int dx = 0; dx < 5; ++dx) {
      int ww = (w + dx + 126) & 127;
      acc += kk[dy * 5 + dx] * vf[hh * 128 + ww];
    }
  }
  int b = f >> 5, t = f & 31;
  vcT[((size_t)(b * 16384 + p)) * 32 + t] = (_Float16)acc;
}

// ---------------- Kernel 5: Gram S = [Q;K][Q;K]^T per batch (WMMA) ----------
__global__ void fno_gram(const _Float16* __restrict__ M, float* __restrict__ S) {
  int b = blockIdx.x;
  int wv = threadIdx.x >> 5;       // i-block
  const _Float16* Mb = M + (size_t)b * 64 * 16384;
  v8f a0 = {}, a1 = {}, a2 = {}, a3 = {};
  for (int kc = 0; kc < 512; ++kc) {
    v16h a  = ld_tile_h16(Mb + (size_t)(wv * 16) * 16384 + kc * 32, 16384);
    v16h b0 = ld_tile_h16(Mb + (size_t)0  * 16384 + kc * 32, 16384);
    v16h b1 = ld_tile_h16(Mb + (size_t)16 * 16384 + kc * 32, 16384);
    v16h b2 = ld_tile_h16(Mb + (size_t)32 * 16384 + kc * 32, 16384);
    v16h b3 = ld_tile_h16(Mb + (size_t)48 * 16384 + kc * 32, 16384);
    a0 = wmma_f16(a, b0, a0);
    a1 = wmma_f16(a, b1, a1);
    a2 = wmma_f16(a, b2, a2);
    a3 = wmma_f16(a, b3, a3);
  }
  float* Sb = S + (size_t)b * 4096;
  st_tile_f32(Sb + wv * 16 * 64 + 0,  64, a0);
  st_tile_f32(Sb + wv * 16 * 64 + 16, 64, a1);
  st_tile_f32(Sb + wv * 16 * 64 + 32, 64, a2);
  st_tile_f32(Sb + wv * 16 * 64 + 48, 64, a3);
}

// ---------------- Kernel 5b: kx = mask * exp(-sqrt(l2^2 + eps)) -------------
__global__ void fno_kx(const float* __restrict__ S, float* __restrict__ kx_out,
                       _Float16* __restrict__ kxA) {
  int b = blockIdx.x, tid = threadIdx.x;  // 1024
  int i = tid >> 5, j = tid & 31;         // i = t_q, j = t_k
  const float* Sb = S + (size_t)b * 4096;
  float d2 = Sb[i * 64 + i] + Sb[(32 + j) * 64 + 32 + j] - 2.f * Sb[i * 64 + 32 + j];
  d2 = fmaxf(d2, 0.f);
  float l2 = sqrtf(d2 + 1e-8f);
  float v = (i == 31) ? 0.f : expf(-l2);
  kx_out[(size_t)b * 1024 + i * 32 + j] = v;
  kxA[(size_t)b * 1024 + j * 32 + i] = (_Float16)v;   // A[q][t] = kx[t][q]
}

// ---------------- Kernel 6: attn_y = kx^T * vc (WMMA), write y-half ---------
__global__ void fno_attn(const _Float16* __restrict__ kxA, const _Float16* __restrict__ vcT,
                         float* __restrict__ out) {
  int pt = blockIdx.x;           // 1024 p-tiles of 16
  int b  = blockIdx.y;
  int qb = threadIdx.x >> 5;     // 2 waves: q-block 0/1
  v16h a  = ld_tile_h16(kxA + (size_t)b * 1024 + qb * 16 * 32, 32);
  v16h bt = ld_tile_h16(vcT + ((size_t)(b * 16384 + pt * 16)) * 32, 32);
  v8f acc = {};
  acc = wmma_f16(a, bt, acc);
  int lane = threadIdx.x & 31;
  int n = lane & 15, mb = (lane & 16) ? 8 : 0;
  int h = (pt * 16) >> 7, w0 = (pt * 16) & 127;
#pragma unroll
  for (int v = 0; v < 8; ++v) {
    int q = qb * 16 + mb + v;
    out[((size_t)((b * 32 + q) * 256 + 128 + h)) * 128 + w0 + n] = acc[v];
  }
}

// ---------------- Launch ----------------------------------------------------
extern "C" void kernel_launch(void* const* d_in, const int* in_sizes, int n_in,
                              void* d_out, int out_size, void* d_ws, size_t ws_size,
                              hipStream_t stream) {
  const float* z    = (const float*)d_in[0];
  const float* qw1r = (const float*)d_in[1];
  const float* qw1i = (const float*)d_in[2];
  const float* qw2r = (const float*)d_in[3];
  const float* qw2i = (const float*)d_in[4];
  const float* kw1r = (const float*)d_in[5];
  const float* kw1i = (const float*)d_in[6];
  const float* kw2r = (const float*)d_in[7];
  const float* kw2i = (const float*)d_in[8];
  const float* icl  = (const float*)d_in[9];
  const float* kyk  = (const float*)d_in[10];
  float* out = (float*)d_out;

  char* ws = (char*)d_ws;
  size_t o = 0;
  auto take = [&](size_t bytes) { char* p = ws + o; o += (bytes + 255) & ~(size_t)255; return p; };
  float*    F1x   = (float*)take(128u * 4096 * 4);
  float*    F1y   = (float*)take(128u * 4096 * 4);
  _Float16* Gq    = (_Float16*)take(128u * 4096 * 2);
  _Float16* Gk    = (_Float16*)take(128u * 4096 * 2);
  _Float16* Gv    = (_Float16*)take(128u * 4096 * 2);
  _Float16* M     = (_Float16*)take((size_t)4 * 64 * 16384 * 2);
  float*    value = (float*)take((size_t)128 * 16384 * 4);
  _Float16* vcT   = (_Float16*)take((size_t)4 * 16384 * 32 * 2);
  float*    S     = (float*)take(4u * 4096 * 4);
  _Float16* kxA   = (_Float16*)take(4u * 1024 * 2);
  _Float16* BwT   = (_Float16*)take(4096u * 2);
  float*    Ec    = (float*)take(4096u * 4);
  float*    Es    = (float*)take(4096u * 4);
  _Float16* BinvT = (_Float16*)take(4096u * 2);

  float* kx_out = out + (size_t)4 * 32 * 256 * 128;

  fno_basis<<<32, 128, 0, stream>>>(BwT, Ec, Es, BinvT);
  fno_zero<<<8192, 256, 0, stream>>>(out);
  fno_fwdw<<<256, 128, 0, stream>>>(z, BwT, F1x, F1y);
  fno_modes<<<128, 256, 0, stream>>>(F1x, F1y, qw1r, qw1i, qw2r, qw2i,
                                     kw1r, kw1i, kw2r, kw2i, icl, Ec, Es, Gq, Gk, Gv);
  fno_invw<<<dim3(128, 3), 128, 0, stream>>>(Gq, Gk, Gv, BinvT, M, value);
  fno_conv<<<8192, 256, 0, stream>>>(value, kyk, vcT);
  fno_gram<<<4, 128, 0, stream>>>(M, S);
  fno_kx<<<4, 1024, 0, stream>>>(S, kx_out, kxA);
  fno_attn<<<dim3(1024, 4), 64, 0, stream>>>(kxA, vcT, out);
}
